// LSTMNew_2834678415414
// MI455X (gfx1250) — matrix operand order repaired and checked by reading, not compiled, not run on hardware
//
#include <hip/hip_runtime.h>

// ---------------- problem constants ----------------
#define LSTM_B 256
#define LSTM_T 1024
#define LSTM_H 512
#define G4H    2048          // 4*H
#define NWG    32            // persistent workgroups, each owns 16 hidden cols
#define TPB    256           // 8 waves of 32
#define LDSROW 72            // 64 halfs + 8 halfs pad -> conflict-free ds_load_b128

typedef __attribute__((ext_vector_type(8)))  _Float16 v8h;
typedef __attribute__((ext_vector_type(16))) _Float16 v16h;
typedef __attribute__((ext_vector_type(8)))  float    v8f;
typedef __attribute__((ext_vector_type(4)))  unsigned int u32x4;
typedef __attribute__((ext_vector_type(4)))  int i32x4;
typedef __attribute__((ext_vector_type(8)))  int i32x8;

// ---------------- workspace layout (bytes) ----------------
constexpr size_t WP1_OFF   = 0;                                      // 2048*512 f16 (Whh1, permuted K-major slices)
constexpr size_t WPI2_OFF  = WP1_OFF  + (size_t)G4H * LSTM_H * 2;    // Wih2
constexpr size_t WPH2_OFF  = WPI2_OFF + (size_t)G4H * LSTM_H * 2;    // Whh2
constexpr size_t B1P_OFF   = WPH2_OFF + (size_t)G4H * LSTM_H * 2;    // 2048 f32 (bih1+bhh1, permuted)
constexpr size_t B2P_OFF   = B1P_OFF  + (size_t)G4H * 4;
constexpr size_t WIH1P_OFF = B2P_OFF  + (size_t)G4H * 4;             // Wih1 column, permuted
constexpr size_t WOUT_OFF  = WIH1P_OFF+ (size_t)G4H * 4;             // 512 f32 (W2@W1)
constexpr size_t BOUT_OFF  = WOUT_OFF + (size_t)LSTM_H * 4;          // 1 f32 (+pad)
constexpr size_t H1_OFF    = BOUT_OFF + 256;                         // 2 buffers * B*H f16
constexpr size_t H2_OFF    = H1_OFF   + 2ull * LSTM_B * LSTM_H * 2;
constexpr size_t C1_OFF    = H2_OFF   + 2ull * LSTM_B * LSTM_H * 2;  // B*H f32
constexpr size_t C2_OFF    = C1_OFF   + (size_t)LSTM_B * LSTM_H * 4;
constexpr size_t CNT_OFF   = C2_OFF   + (size_t)LSTM_B * LSTM_H * 4; // barrier counter
constexpr size_t P_OFF     = CNT_OFF  + 256;                         // NWG * B * T f32 partials
constexpr size_t SMEM_BYTES = 3ull * 512 * LDSROW * 2;               // 216 KB

// ---------------- TDM support ----------------
#if defined(__has_builtin)
#if __has_builtin(__builtin_amdgcn_tensor_load_to_lds)
#define USE_TDM 1
#endif
#endif
#if __has_include(<hip/amd_detail/amd_gfx1250_TDM.h>)
#define TDM_SIX_ARGS 1
#endif

#if defined(USE_TDM)
// Load one 512x64 f16 slice (row-contiguous in global) into LDS with an
// 8-half pad after every 64-half row (pad_interval=32 dwords, pad_amount=4 dwords).
__device__ __forceinline__ void tdm_load_slice(unsigned lds_off, const _Float16* gsrc) {
  unsigned long long ga = (unsigned long long)(uintptr_t)gsrc;
  u32x4 g0;
  g0.x = 1u;                                                  // count=1 (valid user D#)
  g0.y = lds_off;                                             // lds_addr
  g0.z = (unsigned)(ga & 0xFFFFFFFFull);                      // global_addr[31:0]
  g0.w = (unsigned)((ga >> 32) & 0x01FFFFFFull) | 0x80000000u;// global_addr[56:32] | type=2
  i32x8 g1;
  g1[0] = (int)((1u << 16) | (1u << 20) | (4u << 22) | (3u << 25)); // data_size=2B,pad_en,interval=32dw,amount=4dw
  g1[1] = (int)(64u << 16);       // tensor_dim0 = 64
  g1[2] = (int)(512u << 16);      // tensor_dim1 = 512
  g1[3] = (int)(64u << 16);       // tile_dim0 = 64
  g1[4] = 512;                    // tile_dim1 = 512, tile_dim2 = 0
  g1[5] = 64;                     // tensor_dim0_stride = 64
  g1[6] = 0;
  g1[7] = 0;
  i32x4 g2 = {0, 0, 0, 0};
  i32x4 g3 = {0, 0, 0, 0};
#if defined(TDM_SIX_ARGS)
  i32x8 g4 = {0, 0, 0, 0, 0, 0, 0, 0};
  __builtin_amdgcn_tensor_load_to_lds(g0, g1, g2, g3, g4, 0);
#else
  __builtin_amdgcn_tensor_load_to_lds(g0, g1, g2, g3, 0);
#endif
}
#endif

// ---------------- device helpers ----------------
__device__ __forceinline__ float sigf(float x) { return 1.0f / (1.0f + __expf(-x)); }

__device__ __forceinline__ v8f wmma_f16(v16h a, v16h b, v8f c) {
  return __builtin_amdgcn_wmma_f32_16x16x32_f16(false, a, false, b, (short)0, c, false, false);
}

// A fragment (16x32, f16): lanes 0-15 row M=lane, K = k0+{0..7,16..23};
// lanes 16-31 row M=lane-16, K = k0+{8..15,24..31}.
__device__ __forceinline__ v16h load_a(const _Float16* hbase, int rb, int k0, int lane) {
  const int m  = rb + (lane & 15);
  const int kk = k0 + ((lane & 16) ? 8 : 0);
  const _Float16* p = hbase + (size_t)m * LSTM_H + kk;
  v8h lo = *(const v8h*)(p);
  v8h hi = *(const v8h*)(p + 16);
  v16h a;
#pragma unroll
  for (int i = 0; i < 8; ++i) { a[i] = lo[i]; a[i + 8] = hi[i]; }
  return a;
}

// B fragment (32x16, f16) from LDS K-major slice: lane col N=lane%16, K = k0+{0..7,16..23} (lanes 0-15) / k0+{8..15,24..31} (lanes 16-31).
__device__ __forceinline__ v16h load_b(const _Float16* Ws, int k0, int ct, int lane) {
  const int kb = k0 + ((lane & 16) ? 8 : 0);
  const _Float16* p = Ws + (size_t)kb * LDSROW + (ct << 4) + (lane & 15);
  v16h b;
#pragma unroll
  for (int i = 0; i < 8; ++i) { b[i] = p[i * LDSROW]; b[i + 8] = p[(i + 16) * LDSROW]; }
  return b;
}

__device__ __forceinline__ void grid_barrier(unsigned* cnt, unsigned target) {
  __threadfence();
  __syncthreads();
  if (threadIdx.x == 0) {
    __hip_atomic_fetch_add(cnt, 1u, __ATOMIC_RELEASE, __HIP_MEMORY_SCOPE_AGENT);
    while (__hip_atomic_load(cnt, __ATOMIC_ACQUIRE, __HIP_MEMORY_SCOPE_AGENT) < target) {
      __builtin_amdgcn_s_sleep(2);
    }
  }
  __syncthreads();
  __threadfence();
}

// ---------------- prep kernels ----------------
// Permute + convert the three 2048x512 weight matrices to f16, K-major per WG slice:
// wp[wg*32768 + k*64 + gc] = W[j*512 + k], j = (gc/16)*512 + wg*16 + gc%16.
__global__ void prep_weights(const float* __restrict__ Whh1, const float* __restrict__ Wih2,
                             const float* __restrict__ Whh2,
                             _Float16* __restrict__ wp1, _Float16* __restrict__ wpi2,
                             _Float16* __restrict__ wph2) {
  const int i  = blockIdx.x * blockDim.x + threadIdx.x;   // exactly 2048*512 threads
  const int wg = i >> 15;
  const int r  = i & 32767;
  const int k  = r >> 6;
  const int gc = r & 63;
  const int j  = (gc >> 4) * LSTM_H + (wg << 4) + (gc & 15);
  const size_t src = (size_t)j * LSTM_H + k;
  wp1[i]  = (_Float16)Whh1[src];
  wpi2[i] = (_Float16)Wih2[src];
  wph2[i] = (_Float16)Whh2[src];
}

__global__ void prep_small(const float* __restrict__ W1, const float* __restrict__ W2,
                           const float* __restrict__ b1, const float* __restrict__ b2,
                           const float* __restrict__ bih1, const float* __restrict__ bhh1,
                           const float* __restrict__ bih2, const float* __restrict__ bhh2,
                           const float* __restrict__ Wih1,
                           float* __restrict__ b1p, float* __restrict__ b2p,
                           float* __restrict__ wih1p, float* __restrict__ wout,
                           float* __restrict__ bout) {
  const int j = threadIdx.x;   // 512 threads
  float s = 0.f, sb = 0.f;
  for (int k = 0; k < LSTM_H; ++k) {
    const float w2 = W2[k];
    s  += w2 * W1[(size_t)k * LSTM_H + j];
    sb += w2 * b1[k];
  }
  wout[j] = s;
  if (j == 0) bout[0] = sb + b2[0];
  for (int idx = j; idx < G4H; idx += LSTM_H) {
    const int wg = idx >> 6, gc = idx & 63;
    const int jj = (gc >> 4) * LSTM_H + (wg << 4) + (gc & 15);
    b1p[idx]   = bih1[jj] + bhh1[jj];
    b2p[idx]   = bih2[jj] + bhh2[jj];
    wih1p[idx] = Wih1[jj];
  }
}

__global__ void init_state(_Float16* h1s, _Float16* h2s, float* c1, float* c2, unsigned* cnt) {
  const int i = blockIdx.x * blockDim.x + threadIdx.x;
  const int stride = gridDim.x * blockDim.x;
  for (int k = i; k < 2 * LSTM_B * LSTM_H; k += stride) {
    h1s[k] = (_Float16)0.0f;
    h2s[k] = (_Float16)0.0f;
  }
  for (int k = i; k < LSTM_B * LSTM_H; k += stride) {
    c1[k] = 0.0f;
    c2[k] = 0.0f;
  }
  if (i == 0) *cnt = 0u;
}

__global__ void final_reduce(const float* __restrict__ P, const float* __restrict__ bout,
                             float* __restrict__ out) {
  const int i = blockIdx.x * blockDim.x + threadIdx.x;   // B*T threads
  if (i < LSTM_B * LSTM_T) {
    float s = bout[0];
#pragma unroll 1
    for (int w = 0; w < NWG; ++w) s += P[(size_t)w * LSTM_B * LSTM_T + i];
    out[i] = s;
  }
}

// ---------------- main persistent LSTM kernel ----------------
__global__ void __launch_bounds__(TPB, 1)
lstm_persistent(const float* __restrict__ x,
                const _Float16* __restrict__ wp1, const _Float16* __restrict__ wpi2,
                const _Float16* __restrict__ wph2,
                const float* __restrict__ b1p, const float* __restrict__ b2p,
                const float* __restrict__ wih1p, const float* __restrict__ wout,
                _Float16* __restrict__ h1s, _Float16* __restrict__ h2s,
                float* __restrict__ c1, float* __restrict__ c2,
                unsigned* __restrict__ cnt, float* __restrict__ P) {
  extern __shared__ _Float16 smem[];
  _Float16* sW1  = smem;                 // Whh1 slice : 512 x 72 halfs
  _Float16* sWI2 = smem + 512 * LDSROW;  // Wih2 slice
  _Float16* sWH2 = smem + 1024 * LDSROW; // Whh2 slice

  const int wg   = blockIdx.x;
  const int tid  = threadIdx.x;
  const int wave = tid >> 5;
  const int lane = tid & 31;

  // ---- stage this WG's weight slices into LDS (TDM async DMA, padded rows) ----
#if defined(USE_TDM)
  if (wave == 0) {
    tdm_load_slice(0u,      wp1  + (size_t)wg * 32768);
    tdm_load_slice(73728u,  wpi2 + (size_t)wg * 32768);
    tdm_load_slice(147456u, wph2 + (size_t)wg * 32768);
    __builtin_amdgcn_s_wait_tensorcnt(0);
  }
#else
  for (int i = tid; i < 512 * 64; i += TPB) {
    const int k = i >> 6, c = i & 63;
    sW1[k * LDSROW + c]  = wp1[(size_t)wg * 32768 + i];
    sWI2[k * LDSROW + c] = wpi2[(size_t)wg * 32768 + i];
    sWH2[k * LDSROW + c] = wph2[(size_t)wg * 32768 + i];
  }
#endif
  __syncthreads();

  const int n   = lane & 15;              // N within tile == hidden col within WG
  const int mh  = (lane & 16) ? 8 : 0;    // C/D layout: lanes>=16 hold M+8
  const int hc  = (wg << 4) + n;          // global hidden col
  const int rb0 = wave << 5;              // 32 batch rows per wave (two 16-row tiles)
  const int base = wg << 6;

  const float b1_i = b1p[base + n],       b1_f = b1p[base + 16 + n];
  const float b1_g = b1p[base + 32 + n],  b1_o = b1p[base + 48 + n];
  const float w1_i = wih1p[base + n],     w1_f = wih1p[base + 16 + n];
  const float w1_g = wih1p[base + 32 + n],w1_o = wih1p[base + 48 + n];
  const float b2_i = b2p[base + n],       b2_f = b2p[base + 16 + n];
  const float b2_g = b2p[base + 32 + n],  b2_o = b2p[base + 48 + n];
  const float wo = wout[hc];
  const v8f vzero = {0.f, 0.f, 0.f, 0.f, 0.f, 0.f, 0.f, 0.f};
  const size_t BH = (size_t)LSTM_B * LSTM_H;

  unsigned bar = 0;

#pragma unroll 1
  for (int t = 0; t < LSTM_T; ++t) {
    const _Float16* h1c = h1s + (size_t)(t & 1) * BH;         // h1(t-1)
    _Float16*       h1n = h1s + (size_t)((t + 1) & 1) * BH;   // h1(t)
    const _Float16* h2c = h2s + (size_t)(t & 1) * BH;         // h2(t-1)
    _Float16*       h2n = h2s + (size_t)((t + 1) & 1) * BH;   // h2(t)

    // ======== Phase 1: z1 = h1(t-1) @ Whh1^T  (+ x_t*Wih1 + b later) ========
    v8f acc[2][4];
#pragma unroll
    for (int r = 0; r < 2; ++r)
#pragma unroll
      for (int c = 0; c < 4; ++c) acc[r][c] = vzero;

#pragma unroll 2
    for (int kb = 0; kb < 16; ++kb) {
      const int k0 = kb << 5;
      v16h a0 = load_a(h1c, rb0,      k0, lane);
      v16h a1 = load_a(h1c, rb0 + 16, k0, lane);
#pragma unroll
      for (int c = 0; c < 4; ++c) {
        v16h bf = load_b(sW1, k0, c, lane);
        acc[0][c] = wmma_f16(a0, bf, acc[0][c]);
        acc[1][c] = wmma_f16(a1, bf, acc[1][c]);
      }
    }

#pragma unroll
    for (int r = 0; r < 2; ++r) {
#pragma unroll
      for (int v = 0; v < 8; ++v) {
        const int brow = rb0 + (r << 4) + v + mh;
        const float xv = x[(size_t)brow * LSTM_T + t];
        const float zi = acc[r][0][v] + b1_i + xv * w1_i;
        const float zf = acc[r][1][v] + b1_f + xv * w1_f;
        const float zg = acc[r][2][v] + b1_g + xv * w1_g;
        const float zo = acc[r][3][v] + b1_o + xv * w1_o;
        const size_t off = (size_t)brow * LSTM_H + hc;
        const float cn = sigf(zf) * c1[off] + sigf(zi) * tanhf(zg);
        c1[off] = cn;
        h1n[off] = (_Float16)(sigf(zo) * tanhf(cn));
      }
    }

    bar += NWG;
    grid_barrier(cnt, bar);   // h1(t) visible everywhere

    // ======== Phase 2: z2 = h1(t) @ Wih2^T + h2(t-1) @ Whh2^T ========
#pragma unroll
    for (int r = 0; r < 2; ++r)
#pragma unroll
      for (int c = 0; c < 4; ++c) acc[r][c] = vzero;

#pragma unroll 2
    for (int kb = 0; kb < 16; ++kb) {
      const int k0 = kb << 5;
      v16h a0 = load_a(h1n, rb0,      k0, lane);
      v16h a1 = load_a(h1n, rb0 + 16, k0, lane);
      v16h a2 = load_a(h2c, rb0,      k0, lane);
      v16h a3 = load_a(h2c, rb0 + 16, k0, lane);
#pragma unroll
      for (int c = 0; c < 4; ++c) {
        v16h bi = load_b(sWI2, k0, c, lane);
        v16h bh = load_b(sWH2, k0, c, lane);
        acc[0][c] = wmma_f16(a0, bi, acc[0][c]);
        acc[1][c] = wmma_f16(a1, bi, acc[1][c]);
        acc[0][c] = wmma_f16(a2, bh, acc[0][c]);
        acc[1][c] = wmma_f16(a3, bh, acc[1][c]);
      }
    }

#pragma unroll
    for (int r = 0; r < 2; ++r) {
#pragma unroll
      for (int v = 0; v < 8; ++v) {
        const int brow = rb0 + (r << 4) + v + mh;
        const float zi = acc[r][0][v] + b2_i;
        const float zf = acc[r][1][v] + b2_f;
        const float zg = acc[r][2][v] + b2_g;
        const float zo = acc[r][3][v] + b2_o;
        const size_t off = (size_t)brow * LSTM_H + hc;
        const float cn = sigf(zf) * c2[off] + sigf(zi) * tanhf(zg);
        c2[off] = cn;
        const float hn = sigf(zo) * tanhf(cn);
        h2n[off] = (_Float16)hn;
        // partial output: reduce this WG's 16 hidden cols across the half-wave
        float part = hn * wo;
        part += __shfl_xor(part, 1);
        part += __shfl_xor(part, 2);
        part += __shfl_xor(part, 4);
        part += __shfl_xor(part, 8);
        if (n == 0)
          P[(size_t)wg * (LSTM_B * LSTM_T) + (size_t)brow * LSTM_T + t] = part;
      }
    }

    bar += NWG;
    grid_barrier(cnt, bar);   // h2(t) visible; safe to overwrite ping-pong bufs
  }
}

// ---------------- host entry ----------------
extern "C" void kernel_launch(void* const* d_in, const int* in_sizes, int n_in,
                              void* d_out, int out_size, void* d_ws, size_t ws_size,
                              hipStream_t stream) {
  const float* x    = (const float*)d_in[0];
  const float* Wih1 = (const float*)d_in[1];
  const float* Whh1 = (const float*)d_in[2];
  const float* bih1 = (const float*)d_in[3];
  const float* bhh1 = (const float*)d_in[4];
  const float* Wih2 = (const float*)d_in[5];
  const float* Whh2 = (const float*)d_in[6];
  const float* bih2 = (const float*)d_in[7];
  const float* bhh2 = (const float*)d_in[8];
  const float* W1   = (const float*)d_in[9];
  const float* b1   = (const float*)d_in[10];
  const float* W2   = (const float*)d_in[11];
  const float* b2   = (const float*)d_in[12];

  char* ws = (char*)d_ws;
  _Float16* wp1   = (_Float16*)(ws + WP1_OFF);
  _Float16* wpi2  = (_Float16*)(ws + WPI2_OFF);
  _Float16* wph2  = (_Float16*)(ws + WPH2_OFF);
  float*    b1p   = (float*)(ws + B1P_OFF);
  float*    b2p   = (float*)(ws + B2P_OFF);
  float*    wih1p = (float*)(ws + WIH1P_OFF);
  float*    wout  = (float*)(ws + WOUT_OFF);
  float*    bout  = (float*)(ws + BOUT_OFF);
  _Float16* h1s   = (_Float16*)(ws + H1_OFF);
  _Float16* h2s   = (_Float16*)(ws + H2_OFF);
  float*    c1    = (float*)(ws + C1_OFF);
  float*    c2    = (float*)(ws + C2_OFF);
  unsigned* cnt   = (unsigned*)(ws + CNT_OFF);
  float*    P     = (float*)(ws + P_OFF);

  (void)hipFuncSetAttribute(reinterpret_cast<const void*>(lstm_persistent),
                            hipFuncAttributeMaxDynamicSharedMemorySize, (int)SMEM_BYTES);

  prep_weights<<<G4H, LSTM_H, 0, stream>>>(Whh1, Wih2, Whh2, wp1, wpi2, wph2);
  prep_small<<<1, LSTM_H, 0, stream>>>(W1, W2, b1, b2, bih1, bhh1, bih2, bhh2, Wih1,
                                       b1p, b2p, wih1p, wout, bout);
  init_state<<<512, 256, 0, stream>>>(h1s, h2s, c1, c2, cnt);
  lstm_persistent<<<NWG, TPB, SMEM_BYTES, stream>>>(x, wp1, wpi2, wph2, b1p, b2p, wih1p,
                                                    wout, h1s, h2s, c1, c2, cnt, P);
  final_reduce<<<(LSTM_B * LSTM_T + 255) / 256, 256, 0, stream>>>(P, bout, (float*)d_out);
}
